// Model_61083024884235
// MI455X (gfx1250) — compile-verified, matrix operand order checked by reading, not run on hardware
//
#include <hip/hip_runtime.h>
#include <hip/hip_bf16.h>
#include <math.h>

#define D_MODEL 512
#define NH      8
#define DH      64
#define D_FF    2048
#define NB      16
#define LN_EPS  1e-5f
#define BN_EPS  1e-5f

typedef __attribute__((ext_vector_type(16))) __bf16 v16bf;
typedef __attribute__((ext_vector_type(8)))  float  v8f;

union Frag { unsigned u[8]; v16bf v; };

__device__ __forceinline__ unsigned short f2bf(float f) {
  unsigned u = __float_as_uint(f);
  u += 0x7fffu + ((u >> 16) & 1u);          // round to nearest even
  return (unsigned short)(u >> 16);
}

// ---------------------------------------------------------------- GEMM (WMMA)
// C[M,N] = act( A_mapped[M,K] @ W[K,N] + bias )   (optional +=)
// A row mapping: global row gm -> b = gm/Lrow, l = gm%Lrow,
//                src row = b*Lpad + l + rowoff   (handles circular-conv shifts)
// W element (k,n) at W[k*wRow + n*wCol]           (handles conv weight layout)
struct GemmP {
  const float* A; const float* W; const float* bias; float* C;
  int M, N, K, lda;
  int Lrow, Lpad, rowoff;
  int wRow, wCol;
  int gelu, accum;
};

__global__ __launch_bounds__(256) void k_gemm(GemmP p) {
  __shared__ unsigned short As[128 * 32];   // [m][k]  bf16
  __shared__ unsigned short Bs[64 * 32];    // [n][k]  bf16 (transposed)
  const int tid  = threadIdx.x;
  const int wave = tid >> 5, lane = tid & 31, lr = lane & 15, half = lane >> 4;
  const int mBase = blockIdx.x * 128, nBase = blockIdx.y * 64;

  v8f acc[4];
  for (int c = 0; c < 4; ++c)
    for (int r = 0; r < 8; ++r) acc[c][r] = 0.0f;

  for (int k0 = 0; k0 < p.K; k0 += 32) {
    { // stage A tile: 2 threads per row, 16 elems each
      int row = tid >> 1, kk = (tid & 1) * 16;
      int gm = mBase + row;
      unsigned short* dst = &As[row * 32 + kk];
      if (gm < p.M) {
        int b = gm / p.Lrow, l = gm - b * p.Lrow;
        const float* src = p.A + (long)(b * p.Lpad + l + p.rowoff) * p.lda + k0 + kk;
#pragma unroll
        for (int j = 0; j < 16; ++j) dst[j] = f2bf(src[j]);
      } else {
#pragma unroll
        for (int j = 0; j < 16; ++j) dst[j] = 0;
      }
    }
    { // stage B tile (transposed): 4 threads per col, 8 elems each
      int n = tid >> 2, kk = (tid & 3) * 8;
      int gn = nBase + n;
      unsigned short* dst = &Bs[n * 32 + kk];
      if (gn < p.N) {
#pragma unroll
        for (int j = 0; j < 8; ++j)
          dst[j] = f2bf(p.W[(long)(k0 + kk + j) * p.wRow + (long)gn * p.wCol]);
      } else {
#pragma unroll
        for (int j = 0; j < 8; ++j) dst[j] = 0;
      }
    }
    __syncthreads();

    Frag a;
#pragma unroll
    for (int v = 0; v < 8; ++v) {
      int kk = (v < 4 ? 0 : 16) + half * 8 + (v & 3) * 2;   // ISA 16-bit A layout
      a.u[v] = *(const unsigned*)&As[(wave * 16 + lr) * 32 + kk];
    }
#pragma unroll
    for (int c = 0; c < 4; ++c) {
      Frag bm;
#pragma unroll
      for (int v = 0; v < 8; ++v) {
        int kk = (v < 4 ? 0 : 16) + half * 8 + (v & 3) * 2;
        bm.u[v] = *(const unsigned*)&Bs[(c * 16 + lr) * 32 + kk];
      }
      acc[c] = __builtin_amdgcn_wmma_f32_16x16x32_bf16(
          false, a.v, false, bm.v, (short)0, acc[c], false, false);
    }
    __syncthreads();
  }

  // epilogue: C layout lanes0-15 M=r, lanes16-31 M=8+r; N = c*16+lr
#pragma unroll
  for (int c = 0; c < 4; ++c) {
    int gn = nBase + c * 16 + lr;
#pragma unroll
    for (int r = 0; r < 8; ++r) {
      int gm = mBase + wave * 16 + half * 8 + r;
      if (gm < p.M && gn < p.N) {
        float v = acc[c][r];
        if (p.bias) v += p.bias[gn];
        if (p.gelu) v = 0.5f * v * (1.0f + erff(v * 0.70710678118654752f));
        long o = (long)gm * p.N + gn;
        if (p.accum) p.C[o] += v; else p.C[o] = v;
      }
    }
  }
}

// ---------------------------------------------------------------- LayerNorm
// out = LN(x (+ res)) * g + b, one block of 128 threads per 512-wide row
__global__ __launch_bounds__(128) void k_ln(const float* x, const float* res,
                                            const float* g, const float* b,
                                            float* out) {
  __shared__ float red[128];
  const int r = blockIdx.x, tid = threadIdx.x;
  const long base = (long)r * D_MODEL;
  float v[4], s = 0.f;
#pragma unroll
  for (int j = 0; j < 4; ++j) {
    int d = tid + j * 128;
    float t = x[base + d];
    if (res) t += res[base + d];
    v[j] = t; s += t;
  }
  red[tid] = s; __syncthreads();
  for (int st = 64; st > 0; st >>= 1) { if (tid < st) red[tid] += red[tid + st]; __syncthreads(); }
  float mu = red[0] * (1.0f / D_MODEL); __syncthreads();
  float s2 = 0.f;
#pragma unroll
  for (int j = 0; j < 4; ++j) { float d0 = v[j] - mu; s2 += d0 * d0; }
  red[tid] = s2; __syncthreads();
  for (int st = 64; st > 0; st >>= 1) { if (tid < st) red[tid] += red[tid + st]; __syncthreads(); }
  float inv = rsqrtf(red[0] * (1.0f / D_MODEL) + LN_EPS);
#pragma unroll
  for (int j = 0; j < 4; ++j) {
    int d = tid + j * 128;
    out[base + d] = (v[j] - mu) * inv * g[d] + b[d];
  }
}

// ---------------------------------------------------------------- Embedding
// val-circ-conv(Cin->512,k3) + sinusoid PE + mark @ temp_w
__global__ __launch_bounds__(128) void k_embed(const float* x, const float* mark,
    const float* convw, const float* tempw, float* out, int L, int Cin) {
  __shared__ float xs[3 * 8];
  __shared__ float ms[4];
  int bl = blockIdx.x; int b = bl / L, l = bl - b * L;
  int tid = threadIdx.x;
  int lp = (l + L - 1) % L, ln_ = (l + 1) % L;
  if (tid < 3 * Cin) {
    int kk = tid / Cin, c = tid - kk * Cin;
    int ll = (kk == 0) ? lp : (kk == 1 ? l : ln_);
    xs[kk * 8 + c] = x[((long)b * L + ll) * Cin + c];
  }
  if (tid >= 96 && tid < 100) ms[tid - 96] = mark[((long)b * L + l) * 4 + (tid - 96)];
  __syncthreads();
#pragma unroll
  for (int j = 0; j < 4; ++j) {
    int d = tid + j * 128;
    float acc = 0.f;
    for (int c = 0; c < Cin; ++c) {
      const float* w = convw + (long)d * (Cin * 3) + c * 3;
      acc += xs[c] * w[0] + xs[8 + c] * w[1] + xs[16 + c] * w[2];
    }
    int i2 = d >> 1;
    float freq = expf(-(float)(2 * i2) * (9.210340371976184f / 512.0f));
    float arg = (float)l * freq;
    acc += (d & 1) ? cosf(arg) : sinf(arg);
    for (int m = 0; m < 4; ++m) acc += ms[m] * tempw[m * D_MODEL + d];
    out[((long)b * L + l) * D_MODEL + d] = acc;
  }
}

// ---------------------------------------------------------------- ProbSparse
__global__ void k_fill_idx(int* idx, int n, int U, int LK, unsigned seed) {
  int t = blockIdx.x * 256 + threadIdx.x;
  if (t >= n) return; (void)U;
  unsigned x = (unsigned)t * 2654435761u + seed * 0x9E3779B9u;
  x ^= x >> 16; x *= 0x7feb352du; x ^= x >> 15; x *= 0x846ca68bu; x ^= x >> 16;
  idx[t] = (int)(x % (unsigned)LK);
}

// M[b,h,l] = max_u(q.k_sample) - sum_u(q.k_sample)/LK ; one wave per (b,h,l)
__global__ __launch_bounds__(256) void k_probM(const float* q, const float* k,
    const int* idx, float* Mout, int L, int LK, int U, int nWaves) {
  int wv = blockIdx.x * 8 + (threadIdx.x >> 5);
  int lane = threadIdx.x & 31;
  if (wv >= nWaves) return;
  int l = wv % L; int h = (wv / L) % NH; int b = wv / (L * NH);
  long qb = ((long)(b * L + l)) * D_MODEL + h * DH;
  float q0 = q[qb + lane], q1 = q[qb + 32 + lane];
  float mx = -1e38f, sm = 0.f;
  for (int u = 0; u < U; ++u) {
    int kr = idx[l * U + u];
    long kb = ((long)(b * LK + kr)) * D_MODEL + h * DH;
    float pv = q0 * k[kb + lane] + q1 * k[kb + 32 + lane];
    for (int off = 16; off > 0; off >>= 1) pv += __shfl_xor(pv, off, 32);
    mx = fmaxf(mx, pv); sm += pv;
  }
  if (lane == 0) Mout[wv] = mx - sm / (float)LK;
}

__global__ __launch_bounds__(256) void k_topk(const float* M, int* Mtop, int L, int ntop) {
  __shared__ float mv[1024];
  __shared__ float rv[256];
  __shared__ int   ri[256];
  int bh = blockIdx.x, tid = threadIdx.x;
  for (int l = tid; l < L; l += 256) mv[l] = M[(long)bh * L + l];
  __syncthreads();
  for (int t = 0; t < ntop; ++t) {
    float bv = -1e38f; int bi = 0;
    for (int l = tid; l < L; l += 256)
      if (mv[l] > bv) { bv = mv[l]; bi = l; }
    rv[tid] = bv; ri[tid] = bi; __syncthreads();
    for (int st = 128; st > 0; st >>= 1) {
      if (tid < st && rv[tid + st] > rv[tid]) { rv[tid] = rv[tid + st]; ri[tid] = ri[tid + st]; }
      __syncthreads();
    }
    if (tid == 0) { Mtop[bh * ntop + t] = ri[0]; mv[ri[0]] = -1e38f; }
    __syncthreads();
  }
}

__global__ void k_meanv(const float* v, float* mv, int LK) {
  int t = blockIdx.x * 256 + threadIdx.x;       // 16*512 threads
  int b = t >> 9, c = t & 511;
  float s = 0.f;
  for (int l = 0; l < LK; ++l) s += v[((long)(b * LK + l)) * D_MODEL + c];
  mv[t] = s / (float)LK;
}

__global__ void k_bcast(const float* mv, float* ctx, int LQ) {
  long t = (long)blockIdx.x * 256 + threadIdx.x;  // 16*LQ*512
  int c = (int)(t & 511); long bl = t >> 9; int b = (int)(bl / LQ);
  ctx[t] = mv[b * D_MODEL + c];
}

__global__ void k_cumsum(const float* v, float* ctx, int L) {
  int t = blockIdx.x * 256 + threadIdx.x;       // 16*512 threads
  int b = t >> 9, c = t & 511;
  float s = 0.f;
  for (int l = 0; l < L; ++l) {
    long o = ((long)(b * L + l)) * D_MODEL + c;
    s += v[o]; ctx[o] = s;
  }
}

// full score row + softmax + update scatter, one block per (b,h,u)
__global__ __launch_bounds__(256) void k_scores(const float* q, const float* k,
    const float* v, const int* Mtop, float* ctx, int LQ, int LK,
    int ntop, int masked, float scale) {
  __shared__ float sc[1024];
  __shared__ float red[256];
  int u = blockIdx.x % ntop; int h = (blockIdx.x / ntop) % NH; int b = blockIdx.x / (ntop * NH);
  int tid = threadIdx.x;
  int row = Mtop[(b * NH + h) * ntop + u];
  long qb = ((long)(b * LQ + row)) * D_MODEL + h * DH;
  float lmax = -1e38f;
  for (int kk = tid; kk < LK; kk += 256) {
    long kb = ((long)(b * LK + kk)) * D_MODEL + h * DH;
    float dot = 0.f;
    for (int d = 0; d < DH; ++d) dot += q[qb + d] * k[kb + d];
    float s = dot * scale;
    if (masked && kk > row) s = -1e38f;
    sc[kk] = s; lmax = fmaxf(lmax, s);
  }
  red[tid] = lmax; __syncthreads();
  for (int st = 128; st > 0; st >>= 1) { if (tid < st) red[tid] = fmaxf(red[tid], red[tid + st]); __syncthreads(); }
  float mx = red[0]; __syncthreads();
  float lsum = 0.f;
  for (int kk = tid; kk < LK; kk += 256) { float e = expf(sc[kk] - mx); sc[kk] = e; lsum += e; }
  red[tid] = lsum; __syncthreads();
  for (int st = 128; st > 0; st >>= 1) { if (tid < st) red[tid] += red[tid + st]; __syncthreads(); }
  float denom = red[0]; __syncthreads();
  int d = tid & 63, chunk = tid >> 6;
  float acc = 0.f;
  for (int kk = chunk; kk < LK; kk += 4)
    acc += sc[kk] * v[((long)(b * LK + kk)) * D_MODEL + h * DH + d];
  red[tid] = acc; __syncthreads();
  if (tid < 64) {
    float tot = red[tid] + red[tid + 64] + red[tid + 128] + red[tid + 192];
    ctx[((long)(b * LQ + row)) * D_MODEL + h * DH + d] = tot / denom;
  }
}

// ---------------------------------------------------------------- conv stage
__global__ void k_pad(const float* x, float* p, int L, long n) {
  long t = (long)blockIdx.x * 256 + threadIdx.x;
  if (t >= n) return;
  int d = (int)(t & 511); long rest = t >> 9;
  int lp = (int)(rest % (L + 2)); int b = (int)(rest / (L + 2));
  int l = (lp == 0) ? (L - 1) : ((lp == L + 1) ? 0 : lp - 1);
  p[t] = x[((long)(b * L + l)) * D_MODEL + d];
}

__global__ void k_bnelu(float* y, const float* cb, const float* bng,
                        const float* bnb, long n) {
  long t = (long)blockIdx.x * 256 + threadIdx.x;
  if (t >= n) return;
  int d = (int)(t & 511);
  float v = (y[t] + cb[d]) * (bng[d] * rsqrtf(1.0f + BN_EPS)) + bnb[d];
  y[t] = v > 0.f ? v : expm1f(v);
}

__global__ void k_maxpool(const float* y, float* out, int L, long n) {
  long t = (long)blockIdx.x * 256 + threadIdx.x;
  if (t >= n) return;
  int d = (int)(t & 511); long rest = t >> 9;
  int Lo = L >> 1;
  int lo = (int)(rest % Lo); int b = (int)(rest / Lo);
  float m = -1e38f;
  for (int j = 0; j < 3; ++j) {
    int li = 2 * lo - 1 + j;
    if (li >= 0 && li < L) m = fmaxf(m, y[((long)(b * L + li)) * D_MODEL + d]);
  }
  out[t] = m;
}

// ---------------------------------------------------------------- output proj
__global__ void k_proj(const float* x, const float* W, const float* bias,
                       float* out, int L, int PRED, int n) {
  int t = blockIdx.x * 256 + threadIdx.x;
  if (t >= n) return;
  int c = t % 7; int rest = t / 7;
  int lo = rest % PRED; int b = rest / PRED;
  int l = lo + (L - PRED);
  const float* xr = x + ((long)(b * L + l)) * D_MODEL;
  float s = bias[c];
  for (int d = 0; d < D_MODEL; ++d) s += xr[d] * W[d * 7 + c];
  out[t] = s;
}

// ================================================================ host side
struct AttnP { const float *Wq,*bq,*Wk,*bk,*Wv,*bv,*Wo,*bo; };

extern "C" void kernel_launch(void* const* d_in, const int* in_sizes, int n_in,
                              void* d_out, int out_size, void* d_ws, size_t ws_size,
                              hipStream_t stream) {
  (void)in_sizes; (void)n_in; (void)out_size; (void)ws_size;
  int cur = 0;
  auto nx = [&]() -> const float* { return (const float*)d_in[cur++]; };

  const float* x_enc      = nx();
  const float* x_mark_enc = nx();
  const float* x_dec      = nx();
  const float* x_mark_dec = nx();
  const float* enc_conv_w = nx();  const float* enc_temp_w = nx();
  const float* dec_conv_w = nx();  const float* dec_temp_w = nx();

  auto getAttn = [&]() { AttnP a; a.Wq=nx(); a.bq=nx(); a.Wk=nx(); a.bk=nx();
                         a.Wv=nx(); a.bv=nx(); a.Wo=nx(); a.bo=nx(); return a; };
  struct EncL { AttnP attn; const float *W1,*b1,*W2,*b2,*ln1g,*ln1b,*ln2g,*ln2b; } enc[3];
  for (int i = 0; i < 3; ++i) {
    enc[i].attn = getAttn();
    enc[i].W1=nx(); enc[i].b1=nx(); enc[i].W2=nx(); enc[i].b2=nx();
    enc[i].ln1g=nx(); enc[i].ln1b=nx(); enc[i].ln2g=nx(); enc[i].ln2b=nx();
  }
  struct ConvL { const float *w,*b,*bng,*bnb; } cl[2];
  for (int i = 0; i < 2; ++i) { cl[i].w=nx(); cl[i].b=nx(); cl[i].bng=nx(); cl[i].bnb=nx(); }
  const float* encN_g = nx(); const float* encN_b = nx();
  struct DecL { AttnP sa, ca; const float *W1,*b1,*W2,*b2,
                *ln1g,*ln1b,*ln2g,*ln2b,*ln3g,*ln3b; } dec[2];
  for (int j = 0; j < 2; ++j) {
    dec[j].sa = getAttn(); dec[j].ca = getAttn();
    dec[j].W1=nx(); dec[j].b1=nx(); dec[j].W2=nx(); dec[j].b2=nx();
    dec[j].ln1g=nx(); dec[j].ln1b=nx(); dec[j].ln2g=nx(); dec[j].ln2b=nx();
    dec[j].ln3g=nx(); dec[j].ln3b=nx();
  }
  const float* decN_g = nx(); const float* decN_b = nx();
  const float* projW  = nx(); const float* projB  = nx();

  // ---- workspace arenas
  char* ws = (char*)d_ws;
  const size_t SZ = (size_t)NB * 1024 * D_MODEL * sizeof(float);   // 33.5 MB
  float* Hb  = (float*)(ws);             // 4*SZ : FFN hidden / pad buffer
  float* Xb  = (float*)(ws + 4 * SZ);
  float* Yb  = (float*)(ws + 5 * SZ);
  float* Qb  = (float*)(ws + 6 * SZ);
  float* Kb  = (float*)(ws + 7 * SZ);
  float* Vb  = (float*)(ws + 8 * SZ);
  float* Cb  = (float*)(ws + 9 * SZ);
  float* ENC = (float*)(ws + 10 * SZ);
  float* DX  = (float*)(ws + 11 * SZ);
  char*  smp = ws + 12 * SZ;
  float* Mb   = (float*)smp;                      // B*H*1024 f32 = 512 KB
  int*   Mtop = (int*)(smp + (512 << 10));        // 32 KB
  int*   idxb = (int*)(smp + (544 << 10));        // <=144 KB
  float* mvb  = (float*)(smp + (800 << 10));      // B*512 f32 = 32 KB

  auto gemm = [&](const float* A, const float* W, const float* bias, float* C,
                  int M, int N, int K, int lda, int Lrow, int Lpad, int rowoff,
                  int wRow, int wCol, int gelu, int accum) {
    GemmP p{A, W, bias, C, M, N, K, lda, Lrow, Lpad, rowoff, wRow, wCol, gelu, accum};
    dim3 g((M + 127) / 128, (N + 63) / 64);
    k_gemm<<<g, dim3(256), 0, stream>>>(p);
  };
  auto ln = [&](const float* x, const float* res, const float* g,
                const float* b, float* out, int rows) {
    k_ln<<<dim3(rows), dim3(128), 0, stream>>>(x, res, g, b, out);
  };

  auto run_attn = [&](const float* qx, const float* kvx, int LQ, int LK,
                      const AttnP& a, int masked, unsigned seed, float* outY) {
    int Mq = NB * LQ, Mk = NB * LK;
    gemm(qx,  a.Wq, a.bq, Qb, Mq, 512, 512, 512, Mq, Mq, 0, 512, 1, 0, 0);
    gemm(kvx, a.Wk, a.bk, Kb, Mk, 512, 512, 512, Mk, Mk, 0, 512, 1, 0, 0);
    gemm(kvx, a.Wv, a.bv, Vb, Mk, 512, 512, 512, Mk, Mk, 0, 512, 1, 0, 0);
    int U    = (int)fmin(5.0 * ceil(log((double)LK)), (double)LK);
    int ntop = (int)fmin(5.0 * ceil(log((double)LQ)), (double)LQ);
    { int n = LQ * U;
      k_fill_idx<<<dim3((n + 255) / 256), dim3(256), 0, stream>>>(idxb, n, U, LK, seed); }
    { int nw = NB * NH * LQ;
      k_probM<<<dim3(nw / 8), dim3(256), 0, stream>>>(Qb, Kb, idxb, Mb, LQ, LK, U, nw); }
    k_topk<<<dim3(NB * NH), dim3(256), 0, stream>>>(Mb, Mtop, LQ, ntop);
    if (masked) {
      k_cumsum<<<dim3((NB * 512) / 256), dim3(256), 0, stream>>>(Vb, Cb, LK);
    } else {
      k_meanv<<<dim3((NB * 512) / 256), dim3(256), 0, stream>>>(Vb, mvb, LK);
      long n = (long)NB * LQ * 512;
      k_bcast<<<dim3((unsigned)(n / 256)), dim3(256), 0, stream>>>(mvb, Cb, LQ);
    }
    k_scores<<<dim3(NB * NH * ntop), dim3(256), 0, stream>>>(
        Qb, Kb, Vb, Mtop, Cb, LQ, LK, ntop, masked, 0.125f);
    gemm(Cb, a.Wo, a.bo, outY, Mq, 512, 512, 512, Mq, Mq, 0, 512, 1, 0, 0);
  };

  // ================= encoder =================
  k_embed<<<dim3(NB * 1024), dim3(128), 0, stream>>>(
      x_enc, x_mark_enc, enc_conv_w, enc_temp_w, Xb, 1024, 7);
  int L = 1024;
  for (int i = 0; i < 3; ++i) {
    int M = NB * L;
    run_attn(Xb, Xb, L, L, enc[i].attn, 0, (unsigned)i, Yb);
    ln(Xb, Yb, enc[i].ln1g, enc[i].ln1b, Xb, M);
    gemm(Xb, enc[i].W1, enc[i].b1, Hb, M, D_FF, 512, 512, M, M, 0, D_FF, 1, 1, 0);
    gemm(Hb, enc[i].W2, enc[i].b2, Yb, M, 512, D_FF, D_FF, M, M, 0, 512, 1, 0, 0);
    ln(Xb, Yb, enc[i].ln2g, enc[i].ln2b, Xb, M);
    if (i < 2) {
      long np = (long)NB * (L + 2) * 512;
      k_pad<<<dim3((unsigned)((np + 255) / 256)), dim3(256), 0, stream>>>(Xb, Hb, L, np);
      for (int kk = 0; kk < 3; ++kk)
        gemm(Hb, cl[i].w + kk, nullptr, Yb, M, 512, 512, 512,
             L, L + 2, kk, 3, 1536, 0, kk > 0);
      long n = (long)M * 512;
      k_bnelu<<<dim3((unsigned)(n / 256)), dim3(256), 0, stream>>>(
          Yb, cl[i].b, cl[i].bng, cl[i].bnb, n);
      long no = (long)NB * (L / 2) * 512;
      k_maxpool<<<dim3((unsigned)(no / 256)), dim3(256), 0, stream>>>(Yb, Xb, L, no);
      L >>= 1;
    }
  }
  ln(Xb, nullptr, encN_g, encN_b, ENC, NB * L);   // L == 256

  // ================= decoder =================
  k_embed<<<dim3(NB * 1024), dim3(128), 0, stream>>>(
      x_dec, x_mark_dec, dec_conv_w, dec_temp_w, DX, 1024, 7);
  int Md = NB * 1024;
  for (int j = 0; j < 2; ++j) {
    run_attn(DX, DX, 1024, 1024, dec[j].sa, 1, (unsigned)(3 + 2 * j), Yb);
    ln(DX, Yb, dec[j].ln1g, dec[j].ln1b, DX, Md);
    run_attn(DX, ENC, 1024, L, dec[j].ca, 0, (unsigned)(4 + 2 * j), Yb);
    ln(DX, Yb, dec[j].ln2g, dec[j].ln2b, DX, Md);
    gemm(DX, dec[j].W1, dec[j].b1, Hb, Md, D_FF, 512, 512, Md, Md, 0, D_FF, 1, 1, 0);
    gemm(Hb, dec[j].W2, dec[j].b2, Yb, Md, 512, D_FF, D_FF, Md, Md, 0, 512, 1, 0, 0);
    ln(DX, Yb, dec[j].ln3g, dec[j].ln3b, DX, Md);
  }
  ln(DX, nullptr, decN_g, decN_b, DX, Md);
  { int n = NB * 512 * 7;
    k_proj<<<dim3((n + 255) / 256), dim3(256), 0, stream>>>(
        DX, projW, projB, (float*)d_out, 1024, 512, n); }
}
